// Net_modes_50697793962433
// MI455X (gfx1250) — compile-verified, matrix operand order checked by reading, not run on hardware
//
#include <hip/hip_runtime.h>

// ---------------------------------------------------------------------------
// MI455X (gfx1250) fused kernel:
//   s[b,h] = sum_{r,k} W[h,r,k] * x[b,0,k,r];  h = tanh(s);  out = h @ V^T + b
// Flatten c = k*16+r  ->  GEMM [8192 x 8192] x [8192 x 512] (68.7 GFLOP).
// HBM floor 272 MB @ 23.3 TB/s ~ 11.7 us; AI ~252 FLOP/B -> f16 WMMA with f32
// accumulate (f32 16x16x4 ~8x too slow; bf16/fp8 too lossy over K=8192).
// This rev: 64-row blocks, 2Mx4N wave grid (16 wmma / 20 ds_b128 per wave
// per k-slice), register double-buffered global->LDS staging, 2-phase fused
// epilogue (tanh + classifier WMMA + ds_add_f32 cross-wave reduce).
// ---------------------------------------------------------------------------

typedef _Float16 half_t;
typedef __attribute__((ext_vector_type(16))) _Float16 v16h;
typedef __attribute__((ext_vector_type(8)))  _Float16 v8h;
typedef __attribute__((ext_vector_type(4)))  _Float16 v4h;
typedef __attribute__((ext_vector_type(8)))  float    v8f;

#define BDIM   8192
#define KDIM   8192   // R*K = 16*512
#define HIDIM  512
#define NCDIM  16
#define MTILE  64     // batch rows per workgroup
#define KBLK   32     // contraction slice per stage (one WMMA k-step)
#define NITER  (KDIM / KBLK)           // 256 k-slices... (8192/32 = 256)
#define ASTR   40     // LDS row stride (halves): 80B = 5*16B, bank stride 20 dw
#define BSTR   40
#define HSTR   520    // 1040B = 65*16B, bank stride 260 dw

#define SHUF16(lo, hi) __builtin_shufflevector(lo, hi, 0,1,2,3,4,5,6,7,8,9,10,11,12,13,14,15)

// ---------------------------------------------------------------------------
// Prep: W[h][r][k] f32 -> Wt2 f16 in K-blocked layout Wt2[c/32][h][c%32],
// c = k*16+r.  Contiguous reads, in-LDS permute, b128 writes. Wt2 = 8 MB,
// L2-resident for the main GEMM (192 MB L2).
// ---------------------------------------------------------------------------
__global__ __launch_bounds__(256) void prep_w_kernel(const float* __restrict__ W,
                                                     half_t* __restrict__ Wt2) {
    __shared__ __align__(16) half_t tile[KDIM];   // 16 KB
    const int h   = blockIdx.x;
    const int tid = threadIdx.x;
    const float* wrow = W + (size_t)h * KDIM;
    for (int j = 0; j < KDIM / 256; ++j) {
        int idx = j * 256 + tid;               // idx = r*512 + k (contiguous)
        int r = idx >> 9, k = idx & 511;
        tile[(k << 4) | r] = (half_t)wrow[idx];
    }
    __syncthreads();
    for (int j = 0; j < 4; ++j) {
        int chunk = j * 256 + tid;             // 1024 chunks of 8 halves
        int kb = chunk >> 2, cc8 = (chunk & 3) << 3;
        v8h v = *(const v8h*)(tile + chunk * 8);              // ds_read_b128
        *(v8h*)(Wt2 + (size_t)kb * (HIDIM * KBLK) + h * KBLK + cc8) = v;
    }
}

__device__ inline v16h ld_frag(const half_t* base) {   // 2x ds_read_b128
    v8h lo = *(const v8h*)(base);
    v8h hi = *(const v8h*)(base + 16);
    return SHUF16(lo, hi);
}

// ---------------------------------------------------------------------------
// Fused GEMM + tanh + classifier.  WG = 64 batch rows x 512 hidden.
// 8 waves: wm = wave>>2 (32-row half), wn = wave&3 (128-col slab).
// Each wave: acc[2][8] (128 VGPRs), 16 wmma per k-slice.
// ---------------------------------------------------------------------------
__global__ __launch_bounds__(256) void fused_net_kernel(const float* __restrict__ X,
                                                        const half_t* __restrict__ Wt2,
                                                        const float* __restrict__ V,
                                                        const float* __restrict__ bias,
                                                        float* __restrict__ Out) {
    __shared__ __align__(16) half_t At[MTILE * ASTR];     //  5.0 KB
    __shared__ __align__(16) half_t Bt[HIDIM * BSTR];     // 40.0 KB (reused as Ht)
    __shared__ float outS[MTILE * NCDIM];                 //  4.0 KB  (total ~50 KB)

    const int tid   = threadIdx.x;
    const int lane  = tid & 31;
    const int wave  = tid >> 5;
    const int wm    = wave >> 2;            // 0..1 : 32-row half
    const int wn    = wave & 3;             // 0..3 : 128-col slab
    const int b0    = blockIdx.x * MTILE;
    const int l15   = lane & 15;
    const int hi16  = (lane >> 4) << 3;     // 0 / 8  (lane<16 / lane>=16)
    const int khalf = hi16;                 // K-base offset inside frag

    for (int e = tid; e < MTILE * NCDIM; e += 256) outS[e] = 0.0f;

    v8f acc[2][8];
#pragma unroll
    for (int m = 0; m < 2; ++m)
#pragma unroll
        for (int i = 0; i < 8; ++i) acc[m][i] = (v8f){0.f,0.f,0.f,0.f,0.f,0.f,0.f,0.f};

    // Staging registers (double buffer in VGPRs, single LDS buffer).
    // A: 512 float4 / 256 thr = 2 each.  B: 2048 v8h chunks / 256 thr = 8 each.
    const int arow0 = tid >> 3;             // q = tid       -> rows 0..31
    const int arow1 = (256 + tid) >> 3;     // q = 256 + tid -> rows 32..63
    const int ac4   = (tid & 7) << 2;
    float4 aReg[2];
    v8h    bReg[8];

    // Prologue: fetch k-slice 0.
    {
        const float* xb = X + (size_t)b0 * KDIM;
        aReg[0] = *(const float4*)(xb + (size_t)arow0 * KDIM + ac4);
        aReg[1] = *(const float4*)(xb + (size_t)arow1 * KDIM + ac4);
        const half_t* src = Wt2;
#pragma unroll
        for (int j = 0; j < 8; ++j) {
            int chunk = j * 256 + tid;
            bReg[j] = *(const v8h*)(src + (chunk >> 2) * KBLK + ((chunk & 3) << 3));
        }
    }

    for (int kb = 0; kb < NITER; ++kb) {
        __syncthreads();                    // previous slice fully consumed
        // ---- commit staged registers to LDS ----
        {
            v4h h0 = { (half_t)aReg[0].x, (half_t)aReg[0].y, (half_t)aReg[0].z, (half_t)aReg[0].w };
            v4h h1 = { (half_t)aReg[1].x, (half_t)aReg[1].y, (half_t)aReg[1].z, (half_t)aReg[1].w };
            *(v4h*)(At + arow0 * ASTR + ac4) = h0;
            *(v4h*)(At + arow1 * ASTR + ac4) = h1;
#pragma unroll
            for (int j = 0; j < 8; ++j) {
                int chunk = j * 256 + tid;
                *(v8h*)(Bt + (chunk >> 2) * BSTR + ((chunk & 3) << 3)) = bReg[j];
            }
        }
        __syncthreads();
        // ---- prefetch next k-slice into registers (overlaps compute) ----
        if (kb + 1 < NITER) {
            const float* xb = X + (size_t)b0 * KDIM + (kb + 1) * KBLK;
            aReg[0] = *(const float4*)(xb + (size_t)arow0 * KDIM + ac4);
            aReg[1] = *(const float4*)(xb + (size_t)arow1 * KDIM + ac4);
            const half_t* src = Wt2 + (size_t)(kb + 1) * (HIDIM * KBLK);
            __builtin_prefetch(Wt2 + (size_t)(kb + 2) * (HIDIM * KBLK) + tid * 64, 0, 3);
#pragma unroll
            for (int j = 0; j < 8; ++j) {
                int chunk = j * 256 + tid;
                bReg[j] = *(const v8h*)(src + (chunk >> 2) * KBLK + ((chunk & 3) << 3));
            }
        }
        // ---- compute: 16 x v_wmma_f32_16x16x32_f16, B-frag rotated ----
        v16h af0 = ld_frag(At + (wm * 32 +      l15) * ASTR + khalf);
        v16h af1 = ld_frag(At + (wm * 32 + 16 + l15) * ASTR + khalf);
        v16h bf  = ld_frag(Bt + (wn * 128 + l15) * BSTR + khalf);
#pragma unroll
        for (int i = 0; i < 8; ++i) {
            v16h bn;
            if (i < 7) bn = ld_frag(Bt + (wn * 128 + (i + 1) * 16 + l15) * BSTR + khalf);
            acc[0][i] = __builtin_amdgcn_wmma_f32_16x16x32_f16(
                false, af0, false, bf, (short)0, acc[0][i], false, false);
            acc[1][i] = __builtin_amdgcn_wmma_f32_16x16x32_f16(
                false, af1, false, bf, (short)0, acc[1][i], false, false);
            bf = bn;
        }
    }

    // ---- fused epilogue: tanh -> f16 h-tile in LDS (reuse Bt), classifier ----
    __syncthreads();
    half_t* Ht = Bt;                        // 32 x HSTR per phase (33 KB <= 41 KB)
    for (int ph = 0; ph < 2; ++ph) {
        if (wm == ph) {                     // wave-uniform branch: EXEC stays full
#pragma unroll
            for (int mt = 0; mt < 2; ++mt)
#pragma unroll
                for (int i = 0; i < 8; ++i) {
                    int n = wn * 128 + i * 16 + l15;
#pragma unroll
                    for (int v = 0; v < 8; ++v) {     // C layout: M = v + hi16
                        int m = mt * 16 + v + hi16;
                        Ht[m * HSTR + n] = (half_t)tanhf(acc[mt][i][v]);
                    }
                }
        }
        __syncthreads();
        // out[32x16] += Ht[32x512] @ V^T ; wave -> (mi = wave&1, kc = wave>>1)
        {
            int mi = wave & 1;
            int k0 = (wave >> 1) * 128;
            v8f acc2 = (v8f){0.f,0.f,0.f,0.f,0.f,0.f,0.f,0.f};
#pragma unroll
            for (int ks = 0; ks < 4; ++ks) {
                int k = k0 + ks * 32 + khalf;
                v16h af = ld_frag(Ht + (mi * 16 + l15) * HSTR + k);
                // B-frag for V straight from global (L2-resident), f32->f16:
                const float* vr = V + l15 * HIDIM;
                float4 p0 = *(const float4*)(vr + k);
                float4 p1 = *(const float4*)(vr + k + 4);
                float4 p2 = *(const float4*)(vr + k + 16);
                float4 p3 = *(const float4*)(vr + k + 20);
                v16h bf = { (half_t)p0.x,(half_t)p0.y,(half_t)p0.z,(half_t)p0.w,
                            (half_t)p1.x,(half_t)p1.y,(half_t)p1.z,(half_t)p1.w,
                            (half_t)p2.x,(half_t)p2.y,(half_t)p2.z,(half_t)p2.w,
                            (half_t)p3.x,(half_t)p3.y,(half_t)p3.z,(half_t)p3.w };
                acc2 = __builtin_amdgcn_wmma_f32_16x16x32_f16(
                    false, af, false, bf, (short)0, acc2, false, false);
            }
#pragma unroll
            for (int v = 0; v < 8; ++v) {   // cross-wave reduce: ds_add_f32
                int m = ph * 32 + (wave & 1) * 16 + v + hi16;
                atomicAdd(&outS[m * NCDIM + l15], acc2[v]);
            }
        }
        __syncthreads();                    // before next phase overwrites Ht
    }
#pragma unroll
    for (int j = 0; j < 4; ++j) {
        int e = j * 256 + tid;              // 1024 outputs
        int row = e >> 4, cls = e & 15;
        Out[(size_t)(b0 + row) * NCDIM + cls] = outS[e] + bias[cls];
    }
}

// ---------------------------------------------------------------------------
extern "C" void kernel_launch(void* const* d_in, const int* in_sizes, int n_in,
                              void* d_out, int out_size, void* d_ws, size_t ws_size,
                              hipStream_t stream) {
    const float* x  = (const float*)d_in[0];   // [8192, 1, 512, 16]
    const float* W  = (const float*)d_in[1];   // [512, 16, 512]
    const float* V  = (const float*)d_in[2];   // [16, 512]
    const float* b  = (const float*)d_in[3];   // [16]
    float* out      = (float*)d_out;           // [8192, 16]
    half_t* Wt2     = (half_t*)d_ws;           // 8 MB f16 reordered W

    prep_w_kernel<<<HIDIM, 256, 0, stream>>>(W, Wt2);
    fused_net_kernel<<<BDIM / MTILE, 256, 0, stream>>>(x, Wt2, V, b, out);
}
